// MultiHeadSelfAttention_19499151524375
// MI455X (gfx1250) — matrix operand order repaired and checked by reading, not compiled, not run on hardware
//
#include <hip/hip_runtime.h>
#include <hip/hip_bf16.h>

// ---------------------------------------------------------------------------
// MHA forward for MI455X (gfx1250, wave32).  All matmuls on
// v_wmma_f32_16x16x32_bf16 (bf16 operands, f32 accumulate).
//   B=2, S=2048, D_MODEL=1024, H=16, D_H=64
// ---------------------------------------------------------------------------

typedef __attribute__((ext_vector_type(16))) __bf16 v16bf;
typedef __attribute__((ext_vector_type(8)))  __bf16 v8bf;
typedef __attribute__((ext_vector_type(8)))  float  v8f;

#define WMMA_BF16(A,B,C) \
  __builtin_amdgcn_wmma_f32_16x16x32_bf16(false,(A),false,(B),(short)0,(C),false,false)

#define NB   2
#define NS   2048
#define ND   1024
#define NH   16
#define DH   64
#define BS   (NB*NS)          /* 4096 rows */

static __device__ __forceinline__ v8bf ld8(const __bf16* p) { return *(const v8bf*)p; }
static __device__ __forceinline__ v16bf mk16(v8bf lo, v8bf hi) {
  v16bf r;
#pragma unroll
  for (int i = 0; i < 8; ++i) { r[i] = lo[i]; r[i + 8] = hi[i]; }
  return r;
}

// --------------------------- f32 -> bf16 convert ---------------------------
// 8 elements per thread: b128 in (x2), b128 out.
__global__ void cvt_f32_bf16(const float* __restrict__ src,
                             __bf16* __restrict__ dst, int n) {
  int i = (blockIdx.x * blockDim.x + threadIdx.x) * 8;
  if (i + 7 < n) {
    float4 f0 = *(const float4*)(src + i);
    float4 f1 = *(const float4*)(src + i + 4);
    v8bf o;
    o[0] = (__bf16)f0.x; o[1] = (__bf16)f0.y; o[2] = (__bf16)f0.z; o[3] = (__bf16)f0.w;
    o[4] = (__bf16)f1.x; o[5] = (__bf16)f1.y; o[6] = (__bf16)f1.z; o[7] = (__bf16)f1.w;
    *(v8bf*)(dst + i) = o;
  }
}

// ------------------------------ fused QKV GEMM -----------------------------
// out[m][n] = sum_k x[m][k] * w[n][k]  with w = {wq;wk;wv} (N = 3072).
// 8 waves / block; each wave owns a 16(M) x 64(N) tile, software-pipelined
// over K (operands for step k+1 load while WMMAs of step k execute).
// Q,K stored row-major [m][1024]; V stored transposed [b][h][d][s].
__global__ __launch_bounds__(256)
void qkv_gemm(const __bf16* __restrict__ xb, const __bf16* __restrict__ wqkv,
              __bf16* __restrict__ qb, __bf16* __restrict__ kb,
              __bf16* __restrict__ vtb) {
  const int wid  = threadIdx.x >> 5;
  const int lane = threadIdx.x & 31;
  const int hw   = lane >> 4;      // half-wave: 0 or 1
  const int lx   = lane & 15;
  const int m0   = blockIdx.y * 16;
  const int nt   = blockIdx.x * 8 + wid;   // N super-tile in [0,48)
  const int n0   = nt * 64;                // global N in [0,3072)
  const int proj = n0 >> 10;               // 0=Q 1=K 2=V
  const int nin0 = n0 & 1023;
  const __bf16* w = wqkv + (size_t)proj * ND * ND;

  v8f acc[4];
#pragma unroll
  for (int t = 0; t < 4; ++t)
#pragma unroll
    for (int i = 0; i < 8; ++i) acc[t][i] = 0.0f;

  // A operand rows (16 contiguous K per lane; lane>=16: K+16)
  const __bf16* arow = xb + (size_t)(m0 + lx) * ND + hw * 16;
  // B operand rows (16 contiguous K per lane; lane>=16: K+16)
  const __bf16* brow = w + (size_t)(nin0 + lx) * ND + hw * 16;

  v16bf aC = mk16(ld8(arow), ld8(arow + 8));
  v16bf bC[4];
#pragma unroll
  for (int t = 0; t < 4; ++t) {
    const __bf16* pb = brow + (size_t)t * 16 * ND;
    bC[t] = mk16(ld8(pb), ld8(pb + 8));
  }

#pragma unroll
  for (int kk = 0; kk < 31; ++kk) {
    const int k0n = (kk + 1) * 32;
    v16bf aN = mk16(ld8(arow + k0n), ld8(arow + k0n + 8));
    v16bf bN[4];
#pragma unroll
    for (int t = 0; t < 4; ++t) {
      const __bf16* pb = brow + (size_t)t * 16 * ND + k0n;
      bN[t] = mk16(ld8(pb), ld8(pb + 8));
    }
#pragma unroll
    for (int t = 0; t < 4; ++t) acc[t] = WMMA_BF16(aC, bC[t], acc[t]);
    aC = aN;
#pragma unroll
    for (int t = 0; t < 4; ++t) bC[t] = bN[t];
  }
#pragma unroll
  for (int t = 0; t < 4; ++t) acc[t] = WMMA_BF16(aC, bC[t], acc[t]);

  // Epilogue.  Accumulator: lane n<16 VGPR r = D[r][n]; lane n>=16 = D[r+8][n-16].
  if (proj < 2) {
    __bf16* dst = (proj == 0) ? qb : kb;
#pragma unroll
    for (int t = 0; t < 4; ++t) {
      const int col = nin0 + t * 16 + lx;
#pragma unroll
      for (int r = 0; r < 8; ++r)
        dst[(size_t)(m0 + hw * 8 + r) * ND + col] = (__bf16)acc[t][r];
    }
  } else {
    const int b  = m0 >> 11;                 // batch
    const int s0 = (m0 & 2047) + hw * 8;     // seq base (8 contiguous)
#pragma unroll
    for (int t = 0; t < 4; ++t) {
      const int nin = nin0 + t * 16 + lx;
      const int h = nin >> 6, d = nin & 63;
      v8bf v;
#pragma unroll
      for (int r = 0; r < 8; ++r) v[r] = (__bf16)acc[t][r];
      *(v8bf*)(vtb + (size_t)((b * NH + h) * DH + d) * NS + s0) = v;
    }
  }
}

// --------------------------------- RoPE ------------------------------------
// In-place on Q,K (bf16, layout [b][s][h][d]); pairs (2i,2i+1) along d.
__global__ void rope_qk(__bf16* __restrict__ qb, __bf16* __restrict__ kb,
                        const int* __restrict__ pos, int total) {
  int idx = blockIdx.x * blockDim.x + threadIdx.x;
  if (idx >= total) return;
  const int m   = idx >> 9;        // 512 pairs per token (16 heads * 32)
  const int rem = idx & 511;
  const int h   = rem >> 5;
  const int i   = rem & 31;
  const int s   = m & (NS - 1);
  const float p   = (float)pos[s];
  // theta^(-i/32) = exp(-i * ln(10000)/32)
  const float inv = __expf((float)i * -0.28782313662425572f);
  const float ang = p * inv;
  float sn, cs;
  __sincosf(ang, &sn, &cs);
  const size_t off = (size_t)m * ND + h * DH + 2 * i;
  {
    float xe = (float)qb[off], xo = (float)qb[off + 1];
    qb[off]     = (__bf16)(cs * xe - sn * xo);
    qb[off + 1] = (__bf16)(sn * xe + cs * xo);
  }
  {
    float xe = (float)kb[off], xo = (float)kb[off + 1];
    kb[off]     = (__bf16)(cs * xe - sn * xo);
    kb[off + 1] = (__bf16)(sn * xe + cs * xo);
  }
}

// ----------------------- flash attention (transposed) ----------------------
// 8 waves / block; one wave handles 16 queries of one (b,h).
// S^T = K*Q^T (keys x queries), O^T = V^T * P^T (dims x queries);
// online-softmax stats are per-lane scalars.  V^T tile loads are issued
// right after the score WMMAs so their latency hides under the softmax VALU.
__global__ __launch_bounds__(256)
void flash_attn(const __bf16* __restrict__ qb, const __bf16* __restrict__ kb,
                const __bf16* __restrict__ vtb, __bf16* __restrict__ attnb) {
  const int wid  = threadIdx.x >> 5;
  const int lane = threadIdx.x & 31;
  const int hw   = lane >> 4;
  const int lx   = lane & 15;
  const int bh = blockIdx.x * 8 + wid;   // in [0,32)
  const int b  = bh >> 4;
  const int h  = bh & 15;
  const int m0 = blockIdx.y * 16;        // query block

  // Q as B-operands (lane n = query column; 16 contiguous dims per lane)
  const __bf16* qrow = qb + (size_t)(b * NS + m0 + lx) * ND + h * DH;
  const v16bf qB0 = mk16(ld8(qrow + hw * 16),      ld8(qrow + hw * 16 + 8));
  const v16bf qB1 = mk16(ld8(qrow + 32 + hw * 16), ld8(qrow + 32 + hw * 16 + 8));

  v8f accO[4];
#pragma unroll
  for (int t = 0; t < 4; ++t)
#pragma unroll
    for (int i = 0; i < 8; ++i) accO[t][i] = 0.0f;
  float mi = -3.0e38f, li = 0.0f;

  const int row = m0 + lx;             // this lane's query row (both halves)
  for (int j0 = 0; j0 <= m0 + 15; j0 += 32) {   // 32 keys per block
    v8f st0, st1;
#pragma unroll
    for (int i = 0; i < 8; ++i) { st0[i] = 0.0f; st1[i] = 0.0f; }

#pragma unroll
    for (int kt = 0; kt < 2; ++kt) {
      const __bf16* kp = kb + (size_t)(b * NS + j0 + kt * 16 + lx) * ND + h * DH;
      v16bf kA0 = mk16(ld8(kp + hw * 16),      ld8(kp + hw * 16 + 8));
      v16bf kA1 = mk16(ld8(kp + 32 + hw * 16), ld8(kp + 32 + hw * 16 + 8));
      if (kt == 0) { st0 = WMMA_BF16(kA0, qB0, st0); st0 = WMMA_BF16(kA1, qB1, st0); }
      else         { st1 = WMMA_BF16(kA0, qB0, st1); st1 = WMMA_BF16(kA1, qB1, st1); }
    }

    // Issue V^T tile loads now; consumed after the softmax below.
    v16bf vA[4];
#pragma unroll
    for (int t = 0; t < 4; ++t) {
      const __bf16* vp = vtb + (size_t)(bh * DH + t * 16 + lx) * NS + j0 + hw * 16;
      vA[t] = mk16(ld8(vp), ld8(vp + 8));
    }

    // scale + causal mask + tile max   (lane holds keys {j0+hw*8+r, +16})
    float pmax = -3.0e38f;
#pragma unroll
    for (int r = 0; r < 8; ++r) {
      const int key0 = j0 + hw * 8 + r;
      float v0 = st0[r] * 0.125f;            // 1/sqrt(64)
      if (key0 > row) v0 = -3.0e38f;
      float v1 = st1[r] * 0.125f;
      if (key0 + 16 > row) v1 = -3.0e38f;
      st0[r] = v0; st1[r] = v1;
      pmax = fmaxf(pmax, fmaxf(v0, v1));
    }
    pmax = fmaxf(pmax, __shfl_xor(pmax, 16, 32));
    const float mnew  = fmaxf(mi, pmax);
    const float alpha = __expf(mi - mnew);
    float psum = 0.0f;
#pragma unroll
    for (int r = 0; r < 8; ++r) {
      const float p0 = __expf(st0[r] - mnew);
      const float p1 = __expf(st1[r] - mnew);
      st0[r] = p0; st1[r] = p1;
      psum += p0 + p1;
    }
    psum += __shfl_xor(psum, 16, 32);
    li = li * alpha + psum;
    mi = mnew;
#pragma unroll
    for (int t = 0; t < 4; ++t)
#pragma unroll
      for (int i = 0; i < 8; ++i) accO[t][i] *= alpha;

    // P^T accumulator -> B operand: one xor-16 shuffle per element
    v16bf pB;
#pragma unroll
    for (int r = 0; r < 8; ++r) {
      const float o0 = __shfl_xor(st0[r], 16, 32);
      const float o1 = __shfl_xor(st1[r], 16, 32);
      if (hw == 0) { pB[r] = (__bf16)st0[r]; pB[r + 8] = (__bf16)o0; }
      else         { pB[r] = (__bf16)o1;     pB[r + 8] = (__bf16)st1[r]; }
    }

    // O^T += V^T * P^T
#pragma unroll
    for (int t = 0; t < 4; ++t) accO[t] = WMMA_BF16(vA[t], pB, accO[t]);
  }

  // finalize: O^T[d][q]/l ;  lane = query s=m0+lx, VGPRs = 8 contiguous dims
  const float inv_l = 1.0f / li;
  __bf16* orow = attnb + (size_t)(b * NS + m0 + lx) * ND + h * DH;
#pragma unroll
  for (int t = 0; t < 4; ++t) {
    v8bf o;
#pragma unroll
    for (int r = 0; r < 8; ++r) o[r] = (__bf16)(accO[t][r] * inv_l);
    *(v8bf*)(orow + t * 16 + hw * 8) = o;
  }
}

// --------------------------- output projection -----------------------------
// Same software-pipelined structure as qkv_gemm; writes f32 to d_out.
__global__ __launch_bounds__(256)
void oproj_gemm(const __bf16* __restrict__ attnb, const __bf16* __restrict__ wob,
                float* __restrict__ out) {
  const int wid  = threadIdx.x >> 5;
  const int lane = threadIdx.x & 31;
  const int hw   = lane >> 4;
  const int lx   = lane & 15;
  const int m0   = blockIdx.y * 16;
  const int n0   = (blockIdx.x * 8 + wid) * 64;

  v8f acc[4];
#pragma unroll
  for (int t = 0; t < 4; ++t)
#pragma unroll
    for (int i = 0; i < 8; ++i) acc[t][i] = 0.0f;

  const __bf16* arow = attnb + (size_t)(m0 + lx) * ND + hw * 16;
  const __bf16* brow = wob + (size_t)(n0 + lx) * ND + hw * 16;

  v16bf aC = mk16(ld8(arow), ld8(arow + 8));
  v16bf bC[4];
#pragma unroll
  for (int t = 0; t < 4; ++t) {
    const __bf16* pb = brow + (size_t)t * 16 * ND;
    bC[t] = mk16(ld8(pb), ld8(pb + 8));
  }

#pragma unroll
  for (int kk = 0; kk < 31; ++kk) {
    const int k0n = (kk + 1) * 32;
    v16bf aN = mk16(ld8(arow + k0n), ld8(arow + k0n + 8));
    v16bf bN[4];
#pragma unroll
    for (int t = 0; t < 4; ++t) {
      const __bf16* pb = brow + (size_t)t * 16 * ND + k0n;
      bN[t] = mk16(ld8(pb), ld8(pb + 8));
    }
#pragma unroll
    for (int t = 0; t < 4; ++t) acc[t] = WMMA_BF16(aC, bC[t], acc[t]);
    aC = aN;
#pragma unroll
    for (int t = 0; t < 4; ++t) bC[t] = bN[t];
  }
#pragma unroll
  for (int t = 0; t < 4; ++t) acc[t] = WMMA_BF16(aC, bC[t], acc[t]);

#pragma unroll
  for (int t = 0; t < 4; ++t) {
    const int col = n0 + t * 16 + lx;
#pragma unroll
    for (int r = 0; r < 8; ++r)
      out[(size_t)(m0 + hw * 8 + r) * ND + col] = acc[t][r];
  }
}

// ------------------------------- launcher ----------------------------------
extern "C" void kernel_launch(void* const* d_in, const int* in_sizes, int n_in,
                              void* d_out, int out_size, void* d_ws, size_t ws_size,
                              hipStream_t stream) {
  const float* x  = (const float*)d_in[0];
  const float* wq = (const float*)d_in[1];
  const float* wk = (const float*)d_in[2];
  const float* wv = (const float*)d_in[3];
  const float* wo = (const float*)d_in[4];
  const int*  pos = (const int*)d_in[5];

  // workspace layout (bytes): all regions 256-aligned
  char* w8 = (char*)d_ws;
  __bf16* xb    = (__bf16*)(w8);                         // 4194304 elts
  __bf16* wqkv  = (__bf16*)(w8 + 8388608);               // 3 x 1048576
  __bf16* wob   = (__bf16*)(w8 + 8388608 + 6291456);     // 1048576
  __bf16* qb    = (__bf16*)(w8 + 16777216);              // 4194304
  __bf16* kbuf  = (__bf16*)(w8 + 25165824);              // 4194304
  __bf16* vtb   = (__bf16*)(w8 + 33554432);              // 4194304 (transposed)
  __bf16* attnb = (__bf16*)(w8 + 41943040);              // 4194304
  (void)ws_size; (void)in_sizes; (void)n_in; (void)out_size;

  cvt_f32_bf16<<<BS * ND / (256 * 8), 256, 0, stream>>>(x, xb, BS * ND);
  cvt_f32_bf16<<<ND * ND / (256 * 8), 256, 0, stream>>>(wq, wqkv,               ND * ND);
  cvt_f32_bf16<<<ND * ND / (256 * 8), 256, 0, stream>>>(wk, wqkv + 1 * ND * ND, ND * ND);
  cvt_f32_bf16<<<ND * ND / (256 * 8), 256, 0, stream>>>(wv, wqkv + 2 * ND * ND, ND * ND);
  cvt_f32_bf16<<<ND * ND / (256 * 8), 256, 0, stream>>>(wo, wob,                ND * ND);

  qkv_gemm<<<dim3(6, BS / 16), 256, 0, stream>>>(xb, wqkv, qb, kbuf, vtb);
  rope_qk<<<(BS * NH * 32) / 256, 256, 0, stream>>>(qb, kbuf, pos, BS * NH * 32);
  flash_attn<<<dim3(4, NS / 16), 256, 0, stream>>>(qb, kbuf, vtb, attnb);
  oproj_gemm<<<dim3(2, BS / 16), 256, 0, stream>>>(attnb, wob, (float*)d_out);
}
